// TransformerLayer_18253611008557
// MI455X (gfx1250) — matrix operand order repaired and checked by reading, not compiled, not run on hardware
//
#include <hip/hip_runtime.h>
#include <hip/hip_bf16.h>
#include <math.h>

// ---------------- problem constants ----------------
#define B_  4
#define S_  2048
#define H_  256
#define NH_ 8
#define HD_ 32
#define M_  (B_ * S_)          // 8192 token rows
#define EPS_ 1e-5f

// ---------------- WMMA types / helpers ----------------
typedef __attribute__((ext_vector_type(16))) __bf16 bf16x16;
typedef __attribute__((ext_vector_type(8)))  float  v8f;
typedef __attribute__((ext_vector_type(4)))  unsigned int uint32x4;

union FragU { bf16x16 v; unsigned int u[8]; uint32x4 q[2]; };

static __device__ __forceinline__ unsigned short f2bf(float f) {
  return __builtin_bit_cast(unsigned short, (__bf16)f);
}

static __device__ __forceinline__ v8f vzero() {
  v8f z = {0.f, 0.f, 0.f, 0.f, 0.f, 0.f, 0.f, 0.f};
  return z;
}

static __device__ __forceinline__ v8f wmma_bf16(bf16x16 a, bf16x16 b, v8f c) {
  // D = A(16x32 bf16) x B(32x16 bf16) + C(16x16 f32)
  return __builtin_amdgcn_wmma_f32_16x16x32_bf16(
      /*neg_a=*/false, a, /*neg_b=*/false, b,
      /*c_mod=*/(short)0, c, /*reuse_a=*/false, /*reuse_b=*/false);
}

// A fragment: 16x32 bf16, row-major source, rows m=lane&15.
// lanes 0-15 hold K={0..15}, lanes 16-31 hold K={16..31};
// byte offsets per lane: [0..15] and [16..31] -> two b128 loads (16B aligned).
static __device__ __forceinline__ bf16x16
load_afrag(const unsigned short* __restrict__ base, int ld, int lane) {
  FragU f;
  const int m  = lane & 15;
  const int kh = (lane >> 4) << 4;             // +16 for upper half-wave
  const unsigned short* p = base + (size_t)m * ld + kh;
  f.q[0] = *(const uint32x4*)(p);              // K pairs 0,2,4,6 (+kh)
  f.q[1] = *(const uint32x4*)(p + 8);          // K pairs 8,10,12,14 (+kh)
  return f.v;
}

// B fragment where B[k][n] is stored N-major: src[n*ld + k]
// (used for Q*K^T: rows of the K-matrix are key positions -> contiguous in k).
static __device__ __forceinline__ bf16x16
load_bfrag_nmajor(const unsigned short* __restrict__ src, int ld, int k0, int lane) {
  FragU f;
  const int n  = lane & 15;
  const int kb = k0 + ((lane >> 4) << 4);
  const unsigned short* p = src + (size_t)n * ld + kb;
  f.q[0] = *(const uint32x4*)(p);
  f.q[1] = *(const uint32x4*)(p + 8);
  return f.v;
}

// B fragment from pre-packed fragment-major storage: 32 contiguous bytes/lane.
static __device__ __forceinline__ bf16x16
load_bfrag_packed(const unsigned short* __restrict__ p, int lane) {
  FragU f;
  const unsigned short* q = p + (size_t)lane * 16;
  f.q[0] = *(const uint32x4*)(q);
  f.q[1] = *(const uint32x4*)(q + 8);
  return f.v;
}

// ---------------- kernels ----------------

// f32 -> bf16 elementwise
__global__ void cvt_f32_bf16_kernel(const float* __restrict__ src,
                                    unsigned short* __restrict__ dst, int n) {
  for (int i = blockIdx.x * blockDim.x + threadIdx.x; i < n; i += gridDim.x * blockDim.x)
    dst[i] = f2bf(src[i]);
}

// Pack a K-major f32 weight matrix (KxN) into bf16 B-fragment-major tiles:
// dst[((nt16*(K/32) + kblk)*32 + lane)*16 + e], lane=(n%16)+16*((k%32)/16), e=k%16.
__global__ void pack_w_kernel(const float* __restrict__ w,
                              unsigned short* __restrict__ wp, int Kk, int Nn) {
  const int i = blockIdx.x * blockDim.x + threadIdx.x;
  if (i >= Kk * Nn) return;
  const int k = i / Nn, n = i - k * Nn;
  const int nt = n >> 4, nl = n & 15;
  const int kblk = k >> 5, kl = k & 31;
  const int lane = nl + ((kl >> 4) << 4);
  const int e = kl & 15;
  const size_t dst = ((((size_t)nt * (Kk >> 5) + kblk) << 5) + lane) * 16 + e;
  wp[dst] = f2bf(w[i]);
}

// Pack V (bf16, [B,S,H] row-major) into per-(b,h) B-fragment-major tiles:
// dst[((((b*NH+h)*(S/32)+ks)*2 + ntile)*32 + lane)*16 + e]
__global__ void pack_v_kernel(const unsigned short* __restrict__ v,
                              unsigned short* __restrict__ vp) {
  const int i = blockIdx.x * blockDim.x + threadIdx.x;
  if (i >= M_ * H_) return;
  const int rowTok = i >> 8;              // b*S + s  (H=256)
  const int col = i & 255;                // h*32 + c
  const int b = rowTok >> 11, s = rowTok & 2047;
  const int h = col >> 5, c = col & 31;
  const int nt = c >> 4, nl = c & 15;
  const int ks = s >> 5, kl = s & 31;
  const int lane = nl + ((kl >> 4) << 4);
  const int e = kl & 15;
  const size_t dst =
      (((((size_t)(b * NH_ + h) * (S_ >> 5)) + ks) * 2 + nt) * 32 + lane) * 16 + e;
  vp[dst] = v[i];
}

// C = A(bf16 MxK row-major) * Wp(packed bf16 KxN) + bias; per-wave 16x64 tile.
// mode 0: store bf16; mode 1: exact GELU then store bf16; mode 2: store f32.
__global__ void gemm_bf16_kernel(const unsigned short* __restrict__ A,
                                 const unsigned short* __restrict__ Wp,
                                 const float* __restrict__ bias,
                                 unsigned short* __restrict__ outb,
                                 float* __restrict__ outf,
                                 int Mm, int Nn, int Kk, int mode) {
  const int lane = threadIdx.x & 31;
  const int wave = blockIdx.x * (blockDim.x >> 5) + (threadIdx.x >> 5);
  const int ntiles = Nn >> 6;
  const int mt = wave / ntiles;
  const int nt = wave - mt * ntiles;
  if (mt * 16 >= Mm) return;

  v8f acc[4] = {vzero(), vzero(), vzero(), vzero()};
  const unsigned short* arow = A + (size_t)(mt * 16) * Kk;
  const int kblks = Kk >> 5;

  for (int k0 = 0; k0 < Kk; k0 += 32) {
    __builtin_prefetch(arow + (size_t)(lane & 15) * Kk + k0 + 32, 0, 0);
    bf16x16 a = load_afrag(arow + k0, Kk, lane);
#pragma unroll
    for (int j = 0; j < 4; ++j) {
      const unsigned short* wtile =
          Wp + ((((size_t)(nt * 4 + j) * kblks) + (k0 >> 5)) << 5) * 16;
      bf16x16 b = load_bfrag_packed(wtile, lane);
      acc[j] = wmma_bf16(a, b, acc[j]);
    }
  }

  const int ncol  = lane & 15;
  const int rbase = (lane >> 4) << 3;
#pragma unroll
  for (int j = 0; j < 4; ++j) {
    const int n = nt * 64 + j * 16 + ncol;
    const float bv = bias[n];
#pragma unroll
    for (int r = 0; r < 8; ++r) {
      const int row = mt * 16 + rbase + r;
      float v = acc[j][r] + bv;
      if (mode == 1) v = 0.5f * v * (1.0f + erff(v * 0.70710678118654752f));
      const size_t idx = (size_t)row * Nn + n;
      if (mode == 2) outf[idx] = v; else outb[idx] = f2bf(v);
    }
  }
}

// One workgroup (256 thr = 8 waves) per (batch b, 16-row q tile).
// Keeps a 16x2048 score strip and a 16x2048 head-mean strip in LDS (CDNA5: 320KB/WG).
__global__ void attention_kernel(const unsigned short* __restrict__ qg,
                                 const unsigned short* __restrict__ kg,
                                 const unsigned short* __restrict__ vp,
                                 const int* __restrict__ lengths,
                                 float* __restrict__ attnout,
                                 float* __restrict__ atten_mean) {
  extern __shared__ float smem[];
  float* sc     = smem;                 // [16][2048] scores -> probs
  float* mn     = sc + 16 * 2048;       // [16][2048] mean-over-heads accumulator
  float* aux    = mn + 16 * 2048;       // [4096] reductions + P@V partials (8x16x32)
  float* rowmax = aux + 4096;           // [16]
  float* rowsum = rowmax + 16;          // [16]

  const int b    = blockIdx.x >> 7;     // 128 q-tiles per batch
  const int qt   = blockIdx.x & 127;
  const int q0   = qt << 4;
  const int tid  = threadIdx.x;
  const int lane = tid & 31;
  const int wave = tid >> 5;
  const int len  = lengths[b];

  for (int i = tid; i < 16 * 2048; i += 256) mn[i] = 0.f;
  __syncthreads();

  const int row16 = tid >> 4;           // 16 threads per row in reductions
  const int sub16 = tid & 15;

  for (int h = 0; h < NH_; ++h) {
    // ---- Phase 1: S = (Q K^T)/16, causal mask, into LDS ----
    bf16x16 aq = load_afrag(qg + ((size_t)(b * S_ + q0)) * H_ + h * HD_, H_, lane);
    for (int j = wave; j <= qt; j += 8) {
      bf16x16 bkf = load_bfrag_nmajor(
          kg + ((size_t)(b * S_ + j * 16)) * H_ + h * HD_, H_, 0, lane);
      v8f acc = vzero();
      acc = wmma_bf16(aq, bkf, acc);
      const int n    = lane & 15;
      const int kcol = (j << 4) + n;
      const int rb   = (lane >> 4) << 3;
#pragma unroll
      for (int r = 0; r < 8; ++r) {
        const int rr = rb + r;
        float sv = acc[r] * 0.0625f;                 // 1/sqrt(H)=1/16
        if (kcol > q0 + rr) sv = -3.0e38f;           // causal
        sc[rr * 2048 + kcol] = sv;
      }
    }
    __syncthreads();

    // ---- Phase 2: row max and exp-sum over k in [0, qpos] ----
    {
      const int kc = q0 + row16 + 1;
      float m = -3.0e38f;
      for (int k = sub16; k < kc; k += 16) m = fmaxf(m, sc[row16 * 2048 + k]);
      aux[(row16 << 4) + sub16] = m;
    }
    __syncthreads();
    if (tid < 16) {
      float m = aux[tid << 4];
#pragma unroll
      for (int i = 1; i < 16; ++i) m = fmaxf(m, aux[(tid << 4) + i]);
      rowmax[tid] = m;
    }
    __syncthreads();
    {
      const int kc = q0 + row16 + 1;
      const float m = rowmax[row16];
      float s = 0.f;
      for (int k = sub16; k < kc; k += 16) s += __expf(sc[row16 * 2048 + k] - m);
      aux[(row16 << 4) + sub16] = s;
    }
    __syncthreads();
    if (tid < 16) {
      float s = 0.f;
      for (int i = 0; i < 16; ++i) s += aux[(tid << 4) + i];
      rowsum[tid] = s;
    }
    __syncthreads();

    // ---- Phase 3: normalize to probs (0 outside mask), accumulate mean/NH ----
    {
      const int qpos = q0 + row16;
      const bool vr  = (qpos < len) && (h < len);    // row_ok & head_ok
      const float m  = rowmax[row16];
      const float si = vr ? (1.f / rowsum[row16]) : 0.f;
      for (int k = sub16; k < 2048; k += 16) {
        float p = (vr && k <= qpos) ? __expf(sc[row16 * 2048 + k] - m) * si : 0.f;
        sc[row16 * 2048 + k] = p;
        mn[row16 * 2048 + k] += p * 0.125f;          // 1/NH
      }
    }
    __syncthreads();

    // ---- Phase 4: O = P V; each wave owns a 256-wide K slice ----
    v8f o0 = vzero(), o1 = vzero();
    const int ks0 = wave << 8;
    if (ks0 <= q0 + 15) {
      for (int ks = ks0; ks < ks0 + 256; ks += 32) {
        if (ks > q0 + 15) break;                      // wave-uniform
        FragU pf;
        const int mrow = lane & 15;
        const int kh   = (lane >> 4) << 4;
#pragma unroll
        for (int v = 0; v < 8; ++v) {
          const int kk = ((v & 4) << 1) + ((v & 3) << 1) + kh;
          const float2 pp = *(const float2*)&sc[mrow * 2048 + ks + kk];
          pf.u[v] = (unsigned int)f2bf(pp.x) | ((unsigned int)f2bf(pp.y) << 16);
        }
        const unsigned short* vpb =
            vp + (((((size_t)(b * NH_ + h) * (S_ >> 5)) + (ks >> 5)) * 2) << 5) * 16;
        bf16x16 vb0 = load_bfrag_packed(vpb, lane);
        bf16x16 vb1 = load_bfrag_packed(vpb + 32 * 16, lane);
        o0 = wmma_bf16(pf.v, vb0, o0);
        o1 = wmma_bf16(pf.v, vb1, o1);
      }
    }
    {
      const int n  = lane & 15;
      const int rb = (lane >> 4) << 3;
#pragma unroll
      for (int r = 0; r < 8; ++r) {
        aux[((wave << 4) + rb + r) * 32 + n]      = o0[r];
        aux[((wave << 4) + rb + r) * 32 + 16 + n] = o1[r];
      }
    }
    __syncthreads();
    for (int e = tid; e < 512; e += 256) {            // reduce 8 wave partials
      const int r = e >> 5, c = e & 31;
      float s = 0.f;
#pragma unroll
      for (int w = 0; w < 8; ++w) s += aux[((w << 4) + r) * 32 + c];
      attnout[((size_t)(b * S_ + q0 + r)) * H_ + h * HD_ + c] = s;
    }
    __syncthreads();
  }

  // ---- write mean-over-heads strip to output ----
  for (int i = tid; i < 16 * 2048; i += 256) {
    const int r = i >> 11, k = i & 2047;
    atten_mean[((size_t)(b * S_) + q0 + r) * S_ + k] = mn[i];
  }
}

// LayerNorm(a + b) * g + beta ; one block (256 thr) per row of 256 cols.
__global__ void layernorm_kernel(const float* __restrict__ a,
                                 const float* __restrict__ bsrc,
                                 const float* __restrict__ g,
                                 const float* __restrict__ be,
                                 float* __restrict__ out_f32,
                                 unsigned short* __restrict__ out_bf) {
  const int row = blockIdx.x;
  const int c   = threadIdx.x;
  __shared__ float red[256];
  const size_t idx = (size_t)row * H_ + c;
  const float v = a[idx] + bsrc[idx];
  red[c] = v;
  __syncthreads();
  for (int s2 = 128; s2 > 0; s2 >>= 1) {
    if (c < s2) red[c] += red[c + s2];
    __syncthreads();
  }
  const float mu = red[0] * (1.f / H_);
  __syncthreads();
  const float d = v - mu;
  red[c] = d * d;
  __syncthreads();
  for (int s2 = 128; s2 > 0; s2 >>= 1) {
    if (c < s2) red[c] += red[c + s2];
    __syncthreads();
  }
  const float var = red[0] * (1.f / H_);
  const float y = d * rsqrtf(var + EPS_) * g[c] + be[c];
  if (out_f32) out_f32[idx] = y;
  if (out_bf)  out_bf[idx]  = f2bf(y);
}

__global__ void write_lengths_kernel(const int* __restrict__ len, float* __restrict__ out) {
  if (threadIdx.x < B_) out[threadIdx.x] = (float)len[threadIdx.x];
}

// ---------------- launch ----------------
extern "C" void kernel_launch(void* const* d_in, const int* in_sizes, int n_in,
                              void* d_out, int out_size, void* d_ws, size_t ws_size,
                              hipStream_t stream) {
  (void)in_sizes; (void)n_in; (void)out_size; (void)ws_size;

  const float* x        = (const float*)d_in[0];
  const int*   lengths  = (const int*)  d_in[1];
  // d_in[2] = atten_in (unused placeholder)
  const float* Wq = (const float*)d_in[3];  const float* bq = (const float*)d_in[4];
  const float* Wk = (const float*)d_in[5];  const float* bk = (const float*)d_in[6];
  const float* Wv = (const float*)d_in[7];  const float* bv = (const float*)d_in[8];
  const float* W1 = (const float*)d_in[9];  const float* b1 = (const float*)d_in[10];
  const float* W2 = (const float*)d_in[11]; const float* b2 = (const float*)d_in[12];
  const float* g1  = (const float*)d_in[13]; const float* be1 = (const float*)d_in[14];
  const float* g2  = (const float*)d_in[15]; const float* be2 = (const float*)d_in[16];

  float* yout    = (float*)d_out;                       // [B,S,H]
  float* lenout  = yout + (size_t)M_ * H_;              // [B]
  float* meanout = lenout + B_;                         // [B,S,S]

  // workspace carve-out
  char* ws = (char*)d_ws;
  size_t off = 0;
  auto alloc = [&](size_t bytes) -> void* {
    void* p = ws + off;
    off += (bytes + 255) & ~(size_t)255;
    return p;
  };
  unsigned short* xb   = (unsigned short*)alloc((size_t)M_ * H_ * 2);
  unsigned short* wqp  = (unsigned short*)alloc((size_t)H_ * H_ * 2);
  unsigned short* wkp  = (unsigned short*)alloc((size_t)H_ * H_ * 2);
  unsigned short* wvp  = (unsigned short*)alloc((size_t)H_ * H_ * 2);
  unsigned short* w1p  = (unsigned short*)alloc((size_t)H_ * H_ * 2);
  unsigned short* w2p  = (unsigned short*)alloc((size_t)H_ * H_ * 2);
  unsigned short* qb   = (unsigned short*)alloc((size_t)M_ * H_ * 2);
  unsigned short* kb   = (unsigned short*)alloc((size_t)M_ * H_ * 2);
  unsigned short* vbuf = (unsigned short*)alloc((size_t)M_ * H_ * 2);
  unsigned short* vpk  = (unsigned short*)alloc((size_t)M_ * H_ * 2);
  float*          attn = (float*)alloc((size_t)M_ * H_ * 4);
  float*          o1f  = (float*)alloc((size_t)M_ * H_ * 4);   // out1_norm f32
  unsigned short* o1b  = (unsigned short*)alloc((size_t)M_ * H_ * 2);
  unsigned short* h1b  = (unsigned short*)alloc((size_t)M_ * H_ * 2);
  float*          zf   = (float*)alloc((size_t)M_ * H_ * 4);   // FF output pre-LN2

  // 1) convert x to bf16; pack weights (f32 -> bf16 fragment-major)
  cvt_f32_bf16_kernel<<<2048, 256, 0, stream>>>(x, xb, M_ * H_);
  pack_w_kernel<<<256, 256, 0, stream>>>(Wq, wqp, H_, H_);
  pack_w_kernel<<<256, 256, 0, stream>>>(Wk, wkp, H_, H_);
  pack_w_kernel<<<256, 256, 0, stream>>>(Wv, wvp, H_, H_);
  pack_w_kernel<<<256, 256, 0, stream>>>(W1, w1p, H_, H_);
  pack_w_kernel<<<256, 256, 0, stream>>>(W2, w2p, H_, H_);

  // 2) QKV projections (WMMA GEMM, bf16 out)
  const int gemmBlocks = ((M_ / 16) * (H_ / 64) + 7) / 8;   // 8 waves/block
  gemm_bf16_kernel<<<gemmBlocks, 256, 0, stream>>>(xb, wqp, bq, qb,  nullptr, M_, H_, H_, 0);
  gemm_bf16_kernel<<<gemmBlocks, 256, 0, stream>>>(xb, wkp, bk, kb,  nullptr, M_, H_, H_, 0);
  gemm_bf16_kernel<<<gemmBlocks, 256, 0, stream>>>(xb, wvp, bv, vbuf, nullptr, M_, H_, H_, 0);

  // 2b) pack V into fragment-major layout for the P@V WMMA
  pack_v_kernel<<<(M_ * H_ + 255) / 256, 256, 0, stream>>>(vbuf, vpk);

  // 3) attention + atten_mean (big-LDS strip kernel)
  const int SMEM = (16 * 2048 * 2 + 4096 + 32) * (int)sizeof(float);   // ~272 KB (<320 KB/WG)
  (void)hipFuncSetAttribute((const void*)attention_kernel,
                            hipFuncAttributeMaxDynamicSharedMemorySize, SMEM);
  attention_kernel<<<B_ * (S_ / 16), 256, SMEM, stream>>>(qb, kb, vpk, lengths, attn, meanout);

  // 4) LN1 on (x + attn) -> f32 + bf16
  layernorm_kernel<<<M_, 256, 0, stream>>>(x, attn, g1, be1, o1f, o1b);

  // 5) FFN: GELU(out1n @ W1 + b1) @ W2 + b2
  gemm_bf16_kernel<<<gemmBlocks, 256, 0, stream>>>(o1b, w1p, b1, h1b, nullptr, M_, H_, H_, 1);
  gemm_bf16_kernel<<<gemmBlocks, 256, 0, stream>>>(h1b, w2p, b2, nullptr, zf,   M_, H_, H_, 2);

  // 6) LN2 on (out1n + ff) -> y_prim
  layernorm_kernel<<<M_, 256, 0, stream>>>(o1f, zf, g2, be2, yout, nullptr);

  // 7) lengths passthrough
  write_lengths_kernel<<<1, 32, 0, stream>>>(lengths, lenout);
}
